// Modified3DUnet_62466004353135
// MI455X (gfx1250) — matrix operand order repaired and checked by reading, not compiled, not run on hardware
//
#include <hip/hip_runtime.h>
#include <hip/hip_bf16.h>

typedef _Float16 half16 __attribute__((ext_vector_type(16)));
typedef _Float16 half2_t __attribute__((ext_vector_type(2)));
typedef float    float8 __attribute__((ext_vector_type(8)));

#define B_  4
#define N_  2048
#define C_  512
#define G_  32
#define CG_ 16      // channels per group
#define H_  8
#define D_  64
#define M_  (B_ * N_)   // 8192 rows

// ---------------------------------------------------------------------------
// WMMA helpers (CDNA5 v_wmma_f32_16x16x32_f16, layouts per ISA 7.12.2)
// ---------------------------------------------------------------------------
__device__ __forceinline__ float8 wmma_f16(half16 a, half16 b, float8 c) {
    // (neg_a, A, neg_b, B, c_mod, C, reuse_a, reuse_b)
    return __builtin_amdgcn_wmma_f32_16x16x32_f16(false, a, false, b, (short)0, c,
                                                  false, false);
}

// A fragment 16x32: lane holds row m = lane&15; half = lane>>4 selects K groups.
// base must point at &src[row_m * ld + k0] (row-major, K contiguous).
__device__ __forceinline__ half16 load_a_frag(const _Float16* base) {
    const int hf = (threadIdx.x & 31) >> 4;
    half16 a;
#pragma unroll
    for (int i = 0; i < 8; ++i) {
        const int k = ((i >= 4) ? 16 : 0) + hf * 8 + (i & 3) * 2;
        half2_t p = *(const half2_t*)(base + k);
        a[2 * i]     = p.x;
        a[2 * i + 1] = p.y;
    }
    return a;
}

// B fragment 32x16 where memory stores B^T (rows = output column n, K contig).
// base must point at &srcT[col_n * ld + k0].
__device__ __forceinline__ half16 load_b_frag_kcontig(const _Float16* base) {
    const int hf = (threadIdx.x & 31) >> 4;
    half16 b;
#pragma unroll
    for (int i = 0; i < 8; ++i) {
        const int k = ((i >= 4) ? 16 : 0) + hf * 8 + (i & 3) * 2;
        half2_t p = *(const half2_t*)(base + k);
        b[2 * i]     = p.x;
        b[2 * i + 1] = p.y;
    }
    return b;
}

// ---------------------------------------------------------------------------
// Kernel 0: convert f32 weight (K x Ncols) -> f16 transposed (Ncols x K)
// ---------------------------------------------------------------------------
__global__ void wt_transpose_kernel(const float* __restrict__ w,
                                    _Float16* __restrict__ wt,
                                    int K, int Ncols) {
    int idx = blockIdx.x * blockDim.x + threadIdx.x;
    if (idx >= K * Ncols) return;
    int n = idx / K;
    int k = idx - n * K;
    wt[idx] = (_Float16)w[(size_t)k * Ncols + n];
}

// ---------------------------------------------------------------------------
// Kernel 1: GroupNorm statistics per (b, g) over (N, C/G)
// ---------------------------------------------------------------------------
__global__ void gn_stats_kernel(const float* __restrict__ x,
                                float* __restrict__ stats) {
    const int bg = blockIdx.x;             // [0, B*G)
    const int b = bg / G_, g = bg % G_;
    const float* xb = x + (size_t)b * N_ * C_ + g * CG_;
    float s = 0.f, ss = 0.f;
    for (int n = threadIdx.x; n < N_; n += blockDim.x) {
        const float* row = xb + (size_t)n * C_;
#pragma unroll
        for (int c = 0; c < CG_; ++c) {
            float v = row[c];
            s += v;
            ss += v * v;
        }
    }
    __shared__ float red_s[256];
    __shared__ float red_q[256];
    red_s[threadIdx.x] = s;
    red_q[threadIdx.x] = ss;
    __syncthreads();
    for (int off = 128; off > 0; off >>= 1) {
        if ((int)threadIdx.x < off) {
            red_s[threadIdx.x] += red_s[threadIdx.x + off];
            red_q[threadIdx.x] += red_q[threadIdx.x + off];
        }
        __syncthreads();
    }
    if (threadIdx.x == 0) {
        const float cnt = (float)(N_ * CG_);
        float mean = red_s[0] / cnt;
        float var  = red_q[0] / cnt - mean * mean;
        stats[2 * bg]     = mean;
        stats[2 * bg + 1] = rsqrtf(var + 1e-5f);
    }
}

// ---------------------------------------------------------------------------
// Kernel 2: normalize + affine, write h as f16 (M x C)
// ---------------------------------------------------------------------------
__global__ void gn_norm_kernel(const float* __restrict__ x,
                               const float* __restrict__ gscale,
                               const float* __restrict__ gbias,
                               const float* __restrict__ stats,
                               _Float16* __restrict__ h) {
    const int row = blockIdx.x;            // b*N + n
    const int b = row >> 11;               // N_ == 2048
    const float* xr = x + (size_t)row * C_;
    _Float16* hr = h + (size_t)row * C_;
    for (int c = threadIdx.x; c < C_; c += blockDim.x) {
        const int g = c >> 4;              // CG_ == 16
        const float mean = stats[2 * (b * G_ + g)];
        const float rstd = stats[2 * (b * G_ + g) + 1];
        hr[c] = (_Float16)((xr[c] - mean) * rstd * gscale[c] + gbias[c]);
    }
}

// ---------------------------------------------------------------------------
// Kernel 3: QKV GEMM  (M x 512) @ (512 x 1536) -> q,k (B,H,N,d); v (B,H,D,N)
// block = 128 threads = 4 waves; each wave -> 16x128 tile (8 WMMAs per A frag)
// grid (M/16, 1536/512)
// ---------------------------------------------------------------------------
__global__ void qkv_gemm_kernel(const _Float16* __restrict__ h,
                                const _Float16* __restrict__ wt,   // (1536,512)
                                const float* __restrict__ b_qkv,
                                _Float16* __restrict__ q,
                                _Float16* __restrict__ kk,
                                _Float16* __restrict__ v) {
    const int lane = threadIdx.x & 31;
    const int wave = threadIdx.x >> 5;
    const int m0 = blockIdx.x * 16;
    const int n0 = blockIdx.y * 512 + wave * 128;
    const int hf = lane >> 4;
    const _Float16* arow = h + (size_t)(m0 + (lane & 15)) * C_;

    float8 acc[8] = {};
    for (int k0 = 0; k0 < C_; k0 += 32) {
        half16 a = load_a_frag(arow + k0);
#pragma unroll
        for (int t = 0; t < 8; ++t) {
            const int ncol = n0 + t * 16 + (lane & 15);
            acc[t] = wmma_f16(a, load_b_frag_kcontig(wt + (size_t)ncol * C_ + k0),
                              acc[t]);
        }
    }

#pragma unroll
    for (int t = 0; t < 8; ++t) {
        const int col = n0 + t * 16 + (lane & 15);
        const int head = col / 192;        // col / (3*d)
        const int trip = (col / 64) % 3;   // 0=q 1=k 2=v  (uniform within frag)
        const int dd = col & 63;
        const float bias = b_qkv[col];
        if (trip == 2) {
            // V stored transposed: (B, H, D, N) so P@V B-frags are K-contiguous
#pragma unroll
            for (int r = 0; r < 8; ++r) {
                const int m = m0 + r + 8 * hf;
                const int bb = m >> 11;
                const int n = m & (N_ - 1);
                v[(((size_t)bb * H_ + head) * D_ + dd) * N_ + n] =
                    (_Float16)(acc[t][r] + bias);
            }
        } else {
            _Float16* dst = (trip == 0) ? q : kk;
            const float mul = (trip == 0) ? 0.125f : 1.0f;  // fold 1/sqrt(64) in q
#pragma unroll
            for (int r = 0; r < 8; ++r) {
                const int m = m0 + r + 8 * hf;
                const int bb = m >> 11;
                const int n = m & (N_ - 1);
                dst[(((size_t)bb * H_ + head) * N_ + n) * D_ + dd] =
                    (_Float16)((acc[t][r] + bias) * mul);
            }
        }
    }
}

// ---------------------------------------------------------------------------
// Kernel 4: flash attention per (b, h).  block = 256 threads = 8 waves, each
// wave owns a 16-row Q tile; iterate keys in chunks of 32.
// ---------------------------------------------------------------------------
__global__ void attn_kernel(const _Float16* __restrict__ q,
                            const _Float16* __restrict__ k,
                            const _Float16* __restrict__ v,   // (B,H,D,N)
                            _Float16* __restrict__ vals) {
    const int lane = threadIdx.x & 31;
    const int wave = threadIdx.x >> 5;
    const int hf = lane >> 4;
    const int col = lane & 15;
    const int b = blockIdx.z;
    const int hd = blockIdx.y;
    const int q0 = (blockIdx.x * 8 + wave) * 16;

    const size_t base = ((size_t)b * H_ + hd) * N_ * D_;
    const _Float16* Q = q + base;
    const _Float16* K = k + base;
    const _Float16* Vt = v + base;        // (D, N) for this (b,h)

    __shared__ __align__(16) _Float16 pbuf[8][16][32];

    // Q tile: 16 rows x 64 -> two A fragments (k0 = 0, 32)
    const _Float16* qrow = Q + (size_t)(q0 + col) * D_;
    half16 aq0 = load_a_frag(qrow + 0);
    half16 aq1 = load_a_frag(qrow + 32);

    float8 o[4] = {};
    float mrow[8], lrow[8];
#pragma unroll
    for (int r = 0; r < 8; ++r) { mrow[r] = -1e30f; lrow[r] = 0.f; }

    for (int kc = 0; kc < N_; kc += 32) {
        // S = Q @ K^T for 32 keys: two 16x16 score tiles, contraction over d=64
        const _Float16* krow0 = K + (size_t)(kc + col) * D_;
        const _Float16* krow1 = K + (size_t)(kc + 16 + col) * D_;
        float8 s0 = {}, s1 = {};
        s0 = wmma_f16(aq0, load_b_frag_kcontig(krow0 + 0),  s0);
        s0 = wmma_f16(aq1, load_b_frag_kcontig(krow0 + 32), s0);
        s1 = wmma_f16(aq0, load_b_frag_kcontig(krow1 + 0),  s1);
        s1 = wmma_f16(aq1, load_b_frag_kcontig(krow1 + 32), s1);

        // online softmax over the 32 new columns; each lane owns 8 rows
#pragma unroll
        for (int r = 0; r < 8; ++r) {
            float v0 = s0[r], v1 = s1[r];
            float mx = fmaxf(v0, v1);
#pragma unroll
            for (int off = 1; off < 16; off <<= 1)
                mx = fmaxf(mx, __shfl_xor(mx, off, 32));
            const float mnew = fmaxf(mrow[r], mx);
            const float rescale = __expf(mrow[r] - mnew);
            const float p0 = __expf(v0 - mnew);
            const float p1 = __expf(v1 - mnew);
            float ps = p0 + p1;
#pragma unroll
            for (int off = 1; off < 16; off <<= 1)
                ps += __shfl_xor(ps, off, 32);
            lrow[r] = lrow[r] * rescale + ps;
            mrow[r] = mnew;
#pragma unroll
            for (int t = 0; t < 4; ++t) o[t][r] *= rescale;
            const int prow = r + 8 * hf;
            pbuf[wave][prow][col]      = (_Float16)p0;
            pbuf[wave][prow][col + 16] = (_Float16)p1;
        }
        // pbuf slab is wave-private; per-wave DS ops are serviced in order
        // (ISA 7.3: LDS in-order).  wave_barrier() only fences the scheduler.
        __builtin_amdgcn_wave_barrier();
        half16 ap = load_a_frag(&pbuf[wave][col][0]);
        __builtin_amdgcn_wave_barrier();

        // O += P @ V  (16x32 @ 32x64); V transposed -> K-contiguous B frags
#pragma unroll
        for (int t = 0; t < 4; ++t)
            o[t] = wmma_f16(ap,
                            load_b_frag_kcontig(Vt + (size_t)(t * 16 + col) * N_ + kc),
                            o[t]);
    }

    // normalize and scatter into vals (B, N, H*d) row-major f16
#pragma unroll
    for (int t = 0; t < 4; ++t) {
#pragma unroll
        for (int r = 0; r < 8; ++r) {
            const int m = q0 + r + 8 * hf;
            const int dd = t * 16 + col;
            vals[((size_t)b * N_ + m) * C_ + hd * D_ + dd] =
                (_Float16)(o[t][r] / lrow[r]);
        }
    }
}

// ---------------------------------------------------------------------------
// Kernel 5: out = vals @ w_proj + b_proj + x   (f32 output)
// block = 128 threads = 4 waves; each wave -> 16x128 tile; grid (M/16, 1)
// ---------------------------------------------------------------------------
__global__ void proj_gemm_kernel(const _Float16* __restrict__ vals,
                                 const _Float16* __restrict__ wt,  // (512,512)
                                 const float* __restrict__ b_proj,
                                 const float* __restrict__ x,
                                 float* __restrict__ out) {
    const int lane = threadIdx.x & 31;
    const int wave = threadIdx.x >> 5;
    const int m0 = blockIdx.x * 16;
    const int n0 = wave * 128;
    const int hf = lane >> 4;
    const _Float16* arow = vals + (size_t)(m0 + (lane & 15)) * C_;

    float8 acc[8] = {};
    for (int k0 = 0; k0 < C_; k0 += 32) {
        half16 a = load_a_frag(arow + k0);
#pragma unroll
        for (int t = 0; t < 8; ++t) {
            const int ncol = n0 + t * 16 + (lane & 15);
            acc[t] = wmma_f16(a, load_b_frag_kcontig(wt + (size_t)ncol * C_ + k0),
                              acc[t]);
        }
    }

#pragma unroll
    for (int t = 0; t < 8; ++t) {
        const int colIdx = n0 + t * 16 + (lane & 15);
        const float bias = b_proj[colIdx];
#pragma unroll
        for (int r = 0; r < 8; ++r) {
            const size_t m = (size_t)m0 + r + 8 * hf;
            out[m * C_ + colIdx] = acc[t][r] + bias + x[m * C_ + colIdx];
        }
    }
}

// ---------------------------------------------------------------------------
// Host-side launcher
// ---------------------------------------------------------------------------
extern "C" void kernel_launch(void* const* d_in, const int* in_sizes, int n_in,
                              void* d_out, int out_size, void* d_ws, size_t ws_size,
                              hipStream_t stream) {
    const float* x        = (const float*)d_in[0];
    const float* gn_scale = (const float*)d_in[1];
    const float* gn_bias  = (const float*)d_in[2];
    const float* w_qkv    = (const float*)d_in[3];
    const float* b_qkv    = (const float*)d_in[4];
    const float* w_proj   = (const float*)d_in[5];
    const float* b_proj   = (const float*)d_in[6];
    float* out = (float*)d_out;

    // workspace layout (bytes)
    char* ws = (char*)d_ws;
    size_t off = 0;
    float* stats = (float*)(ws + off);            off += 2 * B_ * G_ * sizeof(float);
    off = (off + 255) & ~(size_t)255;
    _Float16* h       = (_Float16*)(ws + off);    off += (size_t)M_ * C_ * 2;
    _Float16* wqkv_t  = (_Float16*)(ws + off);    off += (size_t)3 * C_ * C_ * 2;
    _Float16* wproj_t = (_Float16*)(ws + off);    off += (size_t)C_ * C_ * 2;
    _Float16* qb      = (_Float16*)(ws + off);    off += (size_t)M_ * C_ * 2;
    _Float16* kb      = (_Float16*)(ws + off);    off += (size_t)M_ * C_ * 2;
    _Float16* vb      = (_Float16*)(ws + off);    off += (size_t)M_ * C_ * 2;
    _Float16* valsb   = (_Float16*)(ws + off);    off += (size_t)M_ * C_ * 2;
    (void)ws_size; (void)n_in; (void)in_sizes; (void)out_size;

    // weights -> f16 transposed
    {
        int tot = C_ * 3 * C_;
        wt_transpose_kernel<<<(tot + 255) / 256, 256, 0, stream>>>(w_qkv, wqkv_t,
                                                                   C_, 3 * C_);
    }
    {
        int tot = C_ * C_;
        wt_transpose_kernel<<<(tot + 255) / 256, 256, 0, stream>>>(w_proj, wproj_t,
                                                                   C_, C_);
    }

    // GroupNorm
    gn_stats_kernel<<<B_ * G_, 256, 0, stream>>>(x, stats);
    gn_norm_kernel<<<M_, 256, 0, stream>>>(x, gn_scale, gn_bias, stats, h);

    // QKV projection (v written transposed per-head)
    {
        dim3 grid(M_ / 16, (3 * C_) / 512);
        qkv_gemm_kernel<<<grid, 128, 0, stream>>>(h, wqkv_t, b_qkv, qb, kb, vb);
    }

    // attention
    {
        dim3 grid(N_ / 128, H_, B_);
        attn_kernel<<<grid, 256, 0, stream>>>(qb, kb, vb, valsb);
    }

    // output projection + residual
    {
        dim3 grid(M_ / 16, 1);
        proj_gemm_kernel<<<grid, 128, 0, stream>>>(valsb, wproj_t, b_proj, x, out);
    }
}
